// MultiHeadSelfAttentionWithAttn_33655363731494
// MI455X (gfx1250) — matrix operand order repaired and checked by reading, not compiled, not run on hardware
//
#include <hip/hip_runtime.h>
#include <hip/hip_bf16.h>
#include <stdint.h>

// Problem constants
#define B_    32
#define S_    577
#define SPAD  608              // padded seq (multiple of 32) for K/V tiles
#define E_    768
#define H_    12
#define DH    64
#define BH    (B_*H_)          // 384
#define BS    (B_*S_)          // 18464 (multiple of 32)
#define NQKV  (3*E_)           // 2304
#define QTILES 37              // ceil(577/16)
#define KITERS (SPAD/32)       // 19
#define KBYTES (SPAD*DH*2)     // 77824 bytes: one head's K (or V^T) in bf16
#define PBASE  (2*KBYTES)      // 155648: per-wave P-tile region base in LDS

typedef __bf16 bf16_t;
typedef bf16_t v16bf __attribute__((ext_vector_type(16)));
typedef float  v8f   __attribute__((ext_vector_type(8)));
typedef unsigned int u32x4 __attribute__((ext_vector_type(4)));
typedef int          i32x4 __attribute__((ext_vector_type(4)));
typedef int          i32x8 __attribute__((ext_vector_type(8)));

#define WMMA_BF16(a,b,c) \
  __builtin_amdgcn_wmma_f32_16x16x32_bf16(false,(a),false,(b),(short)0,(c),false,false)

static __device__ __forceinline__ unsigned short f2bf(float f) {
  unsigned int u = __float_as_uint(f);
  u += 0x7FFFu + ((u >> 16) & 1u);   // round-to-nearest-even
  return (unsigned short)(u >> 16);
}

// A fragment: 16x32 bf16, row-major source with leading dim lda.
static __device__ __forceinline__ v16bf ld_fragA(const unsigned short* A, int lda,
                                                 int row0, int k0, int lane) {
  const unsigned short* p = A + (size_t)(row0 + (lane & 15)) * lda + k0 + ((lane >> 4) * 8);
  union { uint4 q[2]; v16bf v; } u;
  u.q[0] = *(const uint4*)(p);
  u.q[1] = *(const uint4*)(p + 16);
  return u.v;
}

// B fragment: 32x16 bf16 taken from B^T stored row-major ([n][k], leading dim ldb).
static __device__ __forceinline__ v16bf ld_fragB(const unsigned short* BT, int ldb,
                                                 int col0, int k0, int lane) {
  const unsigned short* p = BT + (size_t)(col0 + (lane & 15)) * ldb + k0 + ((lane >> 4) * 8);
  union { uint4 q[2]; v16bf v; } u;
  u.q[0] = *(const uint4*)(p);
  u.q[1] = *(const uint4*)(p + 16);
  return u.v;
}

// ---- Tensor Data Mover: 2D tile (dim0 elements/row, dim1 rows) of bf16 -> LDS ----
// Descriptor per CDNA5 ISA (Group0/Group1; groups 2/3 unused for 2D).
static __device__ __forceinline__ void tdm_load_2d(unsigned int lds_off, const void* gptr,
                                                   unsigned int dim0, unsigned int dim1,
                                                   unsigned int stride0) {
  unsigned long long ga = (unsigned long long)(uintptr_t)gptr;
  u32x4 g0;
  g0[0] = 1u;                                             // count=1 valid user descriptor
  g0[1] = lds_off;                                        // lds_addr (bytes)
  g0[2] = (unsigned int)(ga & 0xFFFFFFFFull);             // global_addr[31:0]
  g0[3] = (unsigned int)((ga >> 32) & 0x01FFFFFFull)      // global_addr[56:32]
        | 0x80000000u;                                    // type=2 ("image")
  i32x8 g1;
  g1[0] = 0x00010000;            // workgroup_mask=0 | data_size=1 (2 bytes)
  g1[1] = (int)(dim0 << 16);     // tensor_dim0[15:0] in bits[63:48]
  g1[2] = (int)(dim1 << 16);     // tensor_dim0[31:16]=0 | tensor_dim1[15:0]
  g1[3] = (int)(dim0 << 16);     // tensor_dim1[31:16]=0 | tile_dim0
  g1[4] = (int)dim1;             // tile_dim1 | tile_dim2=0
  g1[5] = (int)stride0;          // tensor_dim0_stride[31:0]
  g1[6] = 0;                     // stride0 hi | tensor_dim1_stride lo (unused, 2D)
  g1[7] = 0;
  i32x4 z4 = {0, 0, 0, 0};
#if defined(__clang_major__) && (__clang_major__ >= 23)
  i32x8 z8 = {0, 0, 0, 0, 0, 0, 0, 0};
  __builtin_amdgcn_tensor_load_to_lds(g0, g1, z4, z4, z8, 0);
#else
  __builtin_amdgcn_tensor_load_to_lds(g0, g1, z4, z4, 0);
#endif
}

// ---------------- prep kernels ----------------
__global__ void k_cvt(const float* __restrict__ in, unsigned short* __restrict__ out,
                      long long n) {
  long long i = (long long)blockIdx.x * blockDim.x + threadIdx.x;
  long long st = (long long)gridDim.x * blockDim.x;
  for (; i < n; i += st) out[i] = f2bf(in[i]);
}

// in: [K][N] f32  ->  out: [N][K] bf16 (transposed)
__global__ void k_transpose_cvt(const float* __restrict__ in, unsigned short* __restrict__ out,
                                int K, int N) {
  long long tot = (long long)K * N;
  long long i = (long long)blockIdx.x * blockDim.x + threadIdx.x;
  long long st = (long long)gridDim.x * blockDim.x;
  for (; i < tot; i += st) {
    int nn = (int)(i / K), kk = (int)(i % K);
    out[i] = f2bf(in[(long long)kk * N + nn]);
  }
}

__global__ void k_zero(uint4* __restrict__ p, long long n4) {
  long long i = (long long)blockIdx.x * blockDim.x + threadIdx.x;
  long long st = (long long)gridDim.x * blockDim.x;
  uint4 z; z.x = z.y = z.z = z.w = 0u;
  for (; i < n4; i += st) p[i] = z;
}

// ---------------- QKV GEMM + bias + 2D RoPE epilogue (wave tile 32x64) ----------------
__global__ void __launch_bounds__(128) k_qkv_gemm(
    const unsigned short* __restrict__ xb, const unsigned short* __restrict__ wqkvT,
    const float* __restrict__ bqkv, unsigned short* __restrict__ Qb,
    unsigned short* __restrict__ Kb, unsigned short* __restrict__ VTb) {
  int lane = threadIdx.x & 31;
  int wid = blockIdx.x * 4 + (threadIdx.x >> 5);
  const int tiles_n = NQKV / 64;                 // 36
  int m0 = (wid / tiles_n) * 32;
  int n0 = (wid % tiles_n) * 64;

  v8f acc[2][4] = { { {}, {}, {}, {} }, { {}, {}, {}, {} } };
  for (int k0 = 0; k0 < E_; k0 += 32) {
    v16bf a0 = ld_fragA(xb, E_, m0,      k0, lane);
    v16bf a1 = ld_fragA(xb, E_, m0 + 16, k0, lane);
    #pragma unroll
    for (int nt = 0; nt < 4; ++nt) {
      v16bf bfr = ld_fragB(wqkvT, E_, n0 + nt * 16, k0, lane);
      acc[0][nt] = WMMA_BF16(a0, bfr, acc[0][nt]);
      acc[1][nt] = WMMA_BF16(a1, bfr, acc[1][nt]);
    }
  }

  int half = lane >> 4, nl = lane & 15;
  for (int mt = 0; mt < 2; ++mt)
    for (int nt = 0; nt < 4; ++nt) {
      #pragma unroll
      for (int r = 0; r < 8; ++r) {
        int Mg = m0 + mt * 16 + r + 8 * half;
        int Ng = n0 + nt * 16 + nl;
        float val = acc[mt][nt][r] + bqkv[Ng];
        float partner = __shfl_xor(val, 1, 32);   // rope pair d ^ 1 (adjacent lane)
        int b = Mg / S_, s = Mg % S_;
        int which = Ng / E_, e = Ng % E_;
        int h = e >> 6, d = e & 63;
        if (which < 2 && s > 0) {                 // RoPE on q,k; token 0 passes through
          int p = s - 1;
          float t = (d < 32) ? (float)(p / 24) : (float)(p % 24);
          int fi = (d & 31) >> 1;
          float ang = t * __expf(-(float)fi * 0.5756462732485114f); // 10000^(-fi/16)
          float sa, ca;
          __sincosf(ang, &sa, &ca);
          float rot = (d & 1) ? partner : -partner;
          val = val * ca + rot * sa;
        }
        unsigned short bits = f2bf(val);
        long long bh = (long long)b * H_ + h;
        if (which == 0)      Qb[(bh * SPAD + s) * DH + d] = bits;
        else if (which == 1) Kb[(bh * SPAD + s) * DH + d] = bits;
        else                 VTb[(bh * DH + d) * SPAD + s] = bits;
      }
    }
}

// ---------------- flash attention: one workgroup per (b,h) ----------------
// TDM stages this head's full K [608x64] and V^T [64x608] (bf16) into LDS once;
// 8 waves then process the 37 query tiles from LDS-resident K/V.
__global__ void __launch_bounds__(256) k_attn(
    const unsigned short* __restrict__ Qb, const unsigned short* __restrict__ Kb,
    const unsigned short* __restrict__ VTb, const unsigned char* __restrict__ kpm,
    unsigned short* __restrict__ ctx) {
  extern __shared__ char lds[];
  int lane = threadIdx.x & 31;
  int w = threadIdx.x >> 5;
  int bh = blockIdx.x;
  int b = bh / H_, h = bh % H_;
  int half = lane >> 4, nl = lane & 15;

  const unsigned short* Qh  = Qb  + (size_t)bh * SPAD * DH;
  const unsigned short* Khg = Kb  + (size_t)bh * SPAD * DH;
  const unsigned short* Vhg = VTb + (size_t)bh * DH * SPAD;

  // DMA K and V^T for this head into LDS (wave 0 issues; EXEC-independent DMA)
  if (w == 0) {
    tdm_load_2d(0u,     Khg, DH,   SPAD, DH);    // K:   608 rows of 64
    tdm_load_2d(KBYTES, Vhg, SPAD, DH,   SPAD);  // V^T: 64 rows of 608
    __builtin_amdgcn_s_wait_tensorcnt(0);
  }
  __syncthreads();

  const unsigned short* Kl = (const unsigned short*)lds;             // [608][64]
  const unsigned short* Vl = (const unsigned short*)(lds + KBYTES);  // [64][608]
  unsigned short* plds = (unsigned short*)(lds + PBASE + w * 1024);  // 16x32 bf16

  const float scale = 0.125f;                    // 1/sqrt(64)

  for (int qt = w; qt < QTILES; qt += 8) {
    int q0 = qt * 16;
    v16bf aq0 = ld_fragA(Qh, DH, q0, 0,  lane);
    v16bf aq1 = ld_fragA(Qh, DH, q0, 32, lane);

    float mI[8], lI[8];
    v8f accv[4] = { {}, {}, {}, {} };
    #pragma unroll
    for (int r = 0; r < 8; ++r) { mI[r] = -1e30f; lI[r] = 0.f; }

    for (int jt = 0; jt < KITERS; ++jt) {
      int kb = jt * 32;
      v8f c0 = {}, c1 = {};
      {
        v16bf bk00 = ld_fragB(Kl, DH, kb,      0,  lane);
        v16bf bk10 = ld_fragB(Kl, DH, kb + 16, 0,  lane);
        c0 = WMMA_BF16(aq0, bk00, c0);
        c1 = WMMA_BF16(aq0, bk10, c1);
        v16bf bk01 = ld_fragB(Kl, DH, kb,      32, lane);
        v16bf bk11 = ld_fragB(Kl, DH, kb + 16, 32, lane);
        c0 = WMMA_BF16(aq1, bk01, c0);
        c1 = WMMA_BF16(aq1, bk11, c1);
      }
      int t0 = kb + nl, t1 = t0 + 16;
      int ti0 = t0 < S_ ? t0 : S_ - 1;
      int ti1 = t1 < S_ ? t1 : S_ - 1;
      bool v0 = (t0 < S_) && (kpm[(size_t)b * S_ + ti0] == 0);
      bool v1 = (t1 < S_) && (kpm[(size_t)b * S_ + ti1] == 0);

      float p0[8], p1[8], alpha[8];
      #pragma unroll
      for (int r = 0; r < 8; ++r) {
        float s0 = v0 ? c0[r] * scale : -1e30f;
        float s1 = v1 ? c1[r] * scale : -1e30f;
        float rmax = fmaxf(s0, s1);
        #pragma unroll
        for (int off = 1; off < 16; off <<= 1)
          rmax = fmaxf(rmax, __shfl_xor(rmax, off, 32));
        float mnew = fmaxf(mI[r], rmax);
        alpha[r] = __expf(mI[r] - mnew);
        mI[r] = mnew;
        p0[r] = __expf(s0 - mnew);
        p1[r] = __expf(s1 - mnew);
        float rs = p0[r] + p1[r];
        #pragma unroll
        for (int off = 1; off < 16; off <<= 1)
          rs += __shfl_xor(rs, off, 32);
        lI[r] = alpha[r] * lI[r] + rs;
      }
      #pragma unroll
      for (int d4 = 0; d4 < 4; ++d4)
        #pragma unroll
        for (int r = 0; r < 8; ++r) accv[d4][r] *= alpha[r];

      // D-layout -> A-layout through per-wave LDS region (DS ops in-order per wave)
      #pragma unroll
      for (int r = 0; r < 8; ++r) {
        int m = r + 8 * half;
        plds[m * 32 + nl]      = f2bf(p0[r]);
        plds[m * 32 + 16 + nl] = f2bf(p1[r]);
      }
      v16bf pf = ld_fragA(plds, 32, 0, 0, lane);

      #pragma unroll
      for (int d4 = 0; d4 < 4; ++d4) {
        v16bf vf = ld_fragB(Vl, SPAD, d4 * 16, kb, lane);
        accv[d4] = WMMA_BF16(pf, vf, accv[d4]);
      }
    }

    #pragma unroll
    for (int r = 0; r < 8; ++r) {
      int m = r + 8 * half;
      int s = q0 + m;
      float inv = 1.0f / lI[r];
      if (s < S_) {
        size_t base = ((size_t)b * S_ + s) * E_ + h * DH;
        #pragma unroll
        for (int d4 = 0; d4 < 4; ++d4)
          ctx[base + d4 * 16 + nl] = f2bf(accv[d4][r] * inv);
      }
    }
  }
}

// ---------------- output projection GEMM + bias, f32 out (wave tile 32x64) ----------------
__global__ void __launch_bounds__(128) k_proj_gemm(
    const unsigned short* __restrict__ ctx, const unsigned short* __restrict__ wprojT,
    const float* __restrict__ bproj, float* __restrict__ out) {
  int lane = threadIdx.x & 31;
  int wid = blockIdx.x * 4 + (threadIdx.x >> 5);
  const int tiles_n = E_ / 64;                   // 12
  int m0 = (wid / tiles_n) * 32;
  int n0 = (wid % tiles_n) * 64;

  v8f acc[2][4] = { { {}, {}, {}, {} }, { {}, {}, {}, {} } };
  for (int k0 = 0; k0 < E_; k0 += 32) {
    v16bf a0 = ld_fragA(ctx, E_, m0,      k0, lane);
    v16bf a1 = ld_fragA(ctx, E_, m0 + 16, k0, lane);
    #pragma unroll
    for (int nt = 0; nt < 4; ++nt) {
      v16bf bfr = ld_fragB(wprojT, E_, n0 + nt * 16, k0, lane);
      acc[0][nt] = WMMA_BF16(a0, bfr, acc[0][nt]);
      acc[1][nt] = WMMA_BF16(a1, bfr, acc[1][nt]);
    }
  }

  int half = lane >> 4, nl = lane & 15;
  for (int mt = 0; mt < 2; ++mt)
    for (int nt = 0; nt < 4; ++nt) {
      #pragma unroll
      for (int r = 0; r < 8; ++r) {
        int Mg = m0 + mt * 16 + r + 8 * half;
        int Ng = n0 + nt * 16 + nl;
        out[(size_t)Mg * E_ + Ng] = acc[mt][nt][r] + bproj[Ng];
      }
    }
}

// ---------------- launcher ----------------
// Workspace layout (bytes), total ~151.1 MB:
//   xb@0, wqkvT@28360704, wprojT@31899648, Qb@33079296, Kb@62963712,
//   VTb@92848128 (transposed [bh][d][token]), ctx@122732544
extern "C" void kernel_launch(void* const* d_in, const int* in_sizes, int n_in,
                              void* d_out, int out_size, void* d_ws, size_t ws_size,
                              hipStream_t stream) {
  const float* x            = (const float*)d_in[0];
  const unsigned char* kpm  = (const unsigned char*)d_in[1];
  const float* W_qkv        = (const float*)d_in[2];
  const float* b_qkv        = (const float*)d_in[3];
  const float* W_proj       = (const float*)d_in[4];
  const float* b_proj       = (const float*)d_in[5];
  float* out                = (float*)d_out;

  char* ws = (char*)d_ws;
  unsigned short* xb     = (unsigned short*)(ws + 0);
  unsigned short* wqkvT  = (unsigned short*)(ws + 28360704);
  unsigned short* wprojT = (unsigned short*)(ws + 31899648);
  unsigned short* Qb     = (unsigned short*)(ws + 33079296);
  unsigned short* Kb     = (unsigned short*)(ws + 62963712);
  unsigned short* VTb    = (unsigned short*)(ws + 92848128);
  unsigned short* ctx    = (unsigned short*)(ws + 122732544);

  k_cvt<<<2048, 256, 0, stream>>>(x, xb, (long long)BS * E_);
  k_transpose_cvt<<<2048, 256, 0, stream>>>(W_qkv, wqkvT, E_, NQKV);
  k_transpose_cvt<<<512, 256, 0, stream>>>(W_proj, wprojT, E_, E_);
  k_zero<<<2048, 256, 0, stream>>>((uint4*)Qb, (long long)(3LL * BH * SPAD * DH * 2 / 16));

  k_qkv_gemm<<<(BS / 32) * (NQKV / 64) / 4, 128, 0, stream>>>(xb, wqkvT, b_qkv, Qb, Kb, VTb);
  k_attn<<<BH, 256, PBASE + 8 * 1024, stream>>>(Qb, Kb, VTb, kpm, ctx);
  k_proj_gemm<<<(BS / 32) * (E_ / 64) / 4, 128, 0, stream>>>(ctx, wprojT, b_proj, out);
}
